// BastileGptOssExperts_15367392985585
// MI455X (gfx1250) — matrix operand order repaired and checked by reading, not compiled, not run on hardware
//
#include <hip/hip_runtime.h>
#include <hip/hip_bf16.h>

// Problem dims (from reference setup_inputs)
#define T_TOK   1024      // B*S tokens
#define H_DIM   1024
#define I_DIM   2048
#define E_EXP   8
#define GU_COLS 4096      // 2*I, interleaved gate/up
#define I_HALF  1024      // act columns per phase

#define TILE_M  32        // tokens per workgroup tile (2 WMMA M-tiles)
#define THREADS 256       // 8 wave32
#define XS      1028      // bf16 elems per X row in LDS (1024+4 pad -> 514 dwords, %64==2)
#define ASH     1028      // bf16 elems per Act-half row (1024+4 pad)
#define SMEM_BYTES (TILE_M*XS*2 + TILE_M*ASH*2 + TILE_M*4 + TILE_M*4)   // ~128.8 KB

#define ALPHA 1.702f
#define LIMIT 7.0f

typedef __attribute__((ext_vector_type(16))) __bf16 v16bf;
typedef __attribute__((ext_vector_type(8)))  float  v8f;

// ---------------------------------------------------------------------------
// A-fragment (16x32 bf16, ISA 7.12.2): lane L -> M = L%16, hi = L/16.
// elems j=0..15 -> K = kk + 16*hi + j (contiguous order used here, not the ISA table; B matches)
// ---------------------------------------------------------------------------
__device__ __forceinline__ v16bf load_a_frag(const __bf16* src, int stride,
                                             int m, int kk, int hi) {
  v16bf a;
  const __bf16* row = src + m * stride + kk + 16 * hi;
#pragma unroll
  for (int j = 0; j < 8; ++j)  a[j] = row[j];
#pragma unroll
  for (int j = 8; j < 16; ++j) a[j] = row[j];
  return a;
}

// ---------------------------------------------------------------------------
// B-fragment (32x16 bf16): lane L -> N = L%16, hi = L/16; elem j -> K = kk+16*hi+j.
// Per-lane gather of fp32 weights (N contiguous -> coalesced across lanes),
// converted to bf16 in-register (v_cvt_pk_bf16_f32).
// ---------------------------------------------------------------------------
__device__ __forceinline__ v16bf load_b_frag(const float* __restrict__ base,
                                             int ldk, int kk, int col, int hi) {
  v16bf b;
  const float* p = base + (size_t)(kk + 16 * hi) * ldk + col;
#pragma unroll
  for (int j = 0; j < 16; ++j) b[j] = (__bf16)p[(size_t)j * ldk];
  return b;
}

__device__ __forceinline__ v8f wmma_bf16(v16bf a, v16bf b, v8f c) {
  return __builtin_amdgcn_wmma_f32_16x16x32_bf16(false, a, false, b,
                                                 (short)0, c, false, false);
}

// ---------------------------------------------------------------------------
// Kernel 0: zero output + per-expert counters (harness poisons buffers).
// ---------------------------------------------------------------------------
__global__ void moe_zero_kernel(float* __restrict__ out, int* __restrict__ cnt) {
  int i = blockIdx.x * blockDim.x + threadIdx.x;
  if (i < T_TOK * H_DIM) out[i] = 0.0f;
  if (i < E_EXP) cnt[i] = 0;
}

// ---------------------------------------------------------------------------
// Kernel 1: routing compaction (dedup of K=2 duplicates == reference sel mask).
// ---------------------------------------------------------------------------
__global__ void moe_route_kernel(const int* __restrict__ ridx,
                                 const float* __restrict__ rw,
                                 int* __restrict__ cnt,
                                 int* __restrict__ toks,
                                 float* __restrict__ wgts) {
  int t = blockIdx.x * blockDim.x + threadIdx.x;
  if (t >= T_TOK) return;
  int e0 = ridx[2 * t + 0];
  int e1 = ridx[2 * t + 1];
  int p = atomicAdd(&cnt[e0], 1);
  toks[e0 * T_TOK + p] = t;
  wgts[e0 * T_TOK + p] = rw[t * E_EXP + e0];
  if (e1 != e0) {
    p = atomicAdd(&cnt[e1], 1);
    toks[e1 * T_TOK + p] = t;
    wgts[e1 * T_TOK + p] = rw[t * E_EXP + e1];
  }
}

// ---------------------------------------------------------------------------
// Kernel 2: fused GEMM1 + GEGLU + GEMM2 per (expert, 32-token tile).
// Two phases over I: act cols [ph*1024, ph*1024+1024).
//   GEMM1(full K=H) -> Act half in LDS -> GEMM2 over that K-slice -> atomics.
// 8 waves; wave w owns act cols w*128..+128 (per phase) and out cols w*128..+128.
// ---------------------------------------------------------------------------
__global__ __launch_bounds__(THREADS)
void moe_fused_kernel(const float* __restrict__ X,
                      const float* __restrict__ GU,
                      const float* __restrict__ GUB,
                      const float* __restrict__ DP,
                      const float* __restrict__ DPB,
                      const int* __restrict__ cnt,
                      const int* __restrict__ toks,
                      const float* __restrict__ wgts,
                      float* __restrict__ out) {
  const int e = blockIdx.y;
  const int nrows = cnt[e];
  const int tile = blockIdx.x;
  if (tile * TILE_M >= nrows) return;   // uniform exit: EXEC stays all-ones

  extern __shared__ char smem[];
  __bf16* Xs    = (__bf16*)smem;                                  // 32 x XS
  __bf16* Act   = (__bf16*)(smem + TILE_M * XS * 2);              // 32 x ASH
  int*    s_tok = (int*)  (smem + TILE_M * XS * 2 + TILE_M * ASH * 2);
  float*  s_wgt = (float*)(s_tok + TILE_M);

  const int tid   = threadIdx.x;
  const int wave  = tid >> 5;
  const int lane  = tid & 31;
  const int laneN = lane & 15;   // N (and M for A-frags)
  const int hi    = lane >> 4;   // lane-half selector

  if (tid < TILE_M) {
    int r = tile * TILE_M + tid;
    if (r < nrows) {
      s_tok[tid] = toks[e * T_TOK + r];
      s_wgt[tid] = wgts[e * T_TOK + r];
    } else {
      s_tok[tid] = -1;
      s_wgt[tid] = 0.0f;
    }
  }
  __syncthreads();

  // ---- Stage X tile (fp32 -> bf16) into LDS, float4-vectorized ----
  for (int c = tid; c < TILE_M * (H_DIM / 4); c += THREADS) {
    int row = c >> 8;            // H_DIM/4 == 256
    int c4  = c & 255;
    int t = s_tok[row];
    float4 v = make_float4(0.f, 0.f, 0.f, 0.f);
    if (t >= 0) v = *(const float4*)(X + (size_t)t * H_DIM + c4 * 4);
    __bf16* d = Xs + row * XS + c4 * 4;
    d[0] = (__bf16)v.x; d[1] = (__bf16)v.y;
    d[2] = (__bf16)v.z; d[3] = (__bf16)v.w;
  }
  __syncthreads();

  const float* GUe = GU + (size_t)e * H_DIM * GU_COLS;
  const float* DPe = DP + (size_t)e * I_DIM * H_DIM;
  const v8f zacc = {0.f, 0.f, 0.f, 0.f, 0.f, 0.f, 0.f, 0.f};

  for (int ph = 0; ph < 2; ++ph) {
    // ---- GEMM1 half + bias + clamp + GEGLU -> Act half (bf16) ----
    for (int tg = 0; tg < 8; tg += 2) {          // 8 N-tiles per wave, 2 at a time
      v8f ag[2][2], au[2][2];                    // [q][m-tile]
#pragma unroll
      for (int q = 0; q < 2; ++q) {
        ag[q][0] = zacc; ag[q][1] = zacc;
        au[q][0] = zacc; au[q][1] = zacc;
      }

      for (int ks = 0; ks < H_DIM / 32; ++ks) {
        int kk = ks * 32;
        v16bf a0 = load_a_frag(Xs,            XS, laneN, kk, hi);  // rows 0..15
        v16bf a1 = load_a_frag(Xs + 16 * XS,  XS, laneN, kk, hi);  // rows 16..31
        if (kk + 32 < H_DIM)   // stream next weight rows toward caches
          __builtin_prefetch(GUe + (size_t)(kk + 32) * GU_COLS
                                 + 2 * (ph * I_HALF + wave * 128 + tg * 16), 0, 0);
#pragma unroll
        for (int q = 0; q < 2; ++q) {
          int icol = ph * I_HALF + wave * 128 + (tg + q) * 16 + laneN;
          v16bf bg = load_b_frag(GUe, GU_COLS, kk, 2 * icol,     hi); // gate: even
          v16bf bu = load_b_frag(GUe, GU_COLS, kk, 2 * icol + 1, hi); // up:   odd
          ag[q][0] = wmma_bf16(a0, bg, ag[q][0]);
          ag[q][1] = wmma_bf16(a1, bg, ag[q][1]);
          au[q][0] = wmma_bf16(a0, bu, au[q][0]);
          au[q][1] = wmma_bf16(a1, bu, au[q][1]);
        }
      }

#pragma unroll
      for (int q = 0; q < 2; ++q) {
        int icl  = wave * 128 + (tg + q) * 16 + laneN;   // local act column
        int icol = ph * I_HALF + icl;                    // global act column
        float bg = GUB[(size_t)e * GU_COLS + 2 * icol];
        float bu = GUB[(size_t)e * GU_COLS + 2 * icol + 1];
#pragma unroll
        for (int mt = 0; mt < 2; ++mt) {
#pragma unroll
          for (int r = 0; r < 8; ++r) {        // C layout: M = r + 8*hi, N = laneN
            float g = ag[q][mt][r] + bg;
            float u = au[q][mt][r] + bu;
            g = fminf(g, LIMIT);
            u = fminf(fmaxf(u, -LIMIT), LIMIT);
            float glu = g * (1.0f / (1.0f + __expf(-ALPHA * g)));
            float av = (u + 1.0f) * glu;
            Act[(mt * 16 + r + 8 * hi) * ASH + icl] = (__bf16)av;
          }
        }
      }
    }
    __syncthreads();   // Act half complete

    // ---- GEMM2 over this K-slice: out += Act[32,1024] x down[k-slice,1024] ----
    for (int tg = 0; tg < 8; tg += 4) {
      v8f acc[4][2];
#pragma unroll
      for (int q = 0; q < 4; ++q) { acc[q][0] = zacc; acc[q][1] = zacc; }

      for (int ks = 0; ks < I_HALF / 32; ++ks) {
        int kloc = ks * 32;
        v16bf a0 = load_a_frag(Act,           ASH, laneN, kloc, hi);
        v16bf a1 = load_a_frag(Act + 16 * ASH, ASH, laneN, kloc, hi);
        if (kloc + 32 < I_HALF)
          __builtin_prefetch(DPe + (size_t)(ph * I_HALF + kloc + 32) * H_DIM
                                 + wave * 128 + tg * 16, 0, 0);
#pragma unroll
        for (int q = 0; q < 4; ++q) {
          int hcol = wave * 128 + (tg + q) * 16 + laneN;
          v16bf b = load_b_frag(DPe, H_DIM, ph * I_HALF + kloc, hcol, hi);
          acc[q][0] = wmma_bf16(a0, b, acc[q][0]);
          acc[q][1] = wmma_bf16(a1, b, acc[q][1]);
        }
      }

#pragma unroll
      for (int q = 0; q < 4; ++q) {
        int hcol = wave * 128 + (tg + q) * 16 + laneN;
        float bias = (ph == 0) ? DPB[(size_t)e * H_DIM + hcol] : 0.0f;
#pragma unroll
        for (int mt = 0; mt < 2; ++mt) {
#pragma unroll
          for (int r = 0; r < 8; ++r) {
            int m = mt * 16 + r + 8 * hi;
            int t = s_tok[m];
            if (t >= 0)
              atomicAdd(out + (size_t)t * H_DIM + hcol,
                        (acc[q][mt][r] + bias) * s_wgt[m]);
          }
        }
      }
    }
    __syncthreads();   // before Act is overwritten by next phase
  }
}

// ---------------------------------------------------------------------------
extern "C" void kernel_launch(void* const* d_in, const int* in_sizes, int n_in,
                              void* d_out, int out_size, void* d_ws, size_t ws_size,
                              hipStream_t stream) {
  const float* X   = (const float*)d_in[0];  // hidden_states [2,512,1024]
  const int*   RI  = (const int*)  d_in[1];  // router_indices [1024,2]
  const float* RW  = (const float*)d_in[2];  // routing_weights [1024,8]
  const float* GU  = (const float*)d_in[3];  // gate_up_proj [8,1024,4096]
  const float* GUB = (const float*)d_in[4];  // gate_up_proj_bias [8,4096]
  const float* DP  = (const float*)d_in[5];  // down_proj [8,2048,1024]
  const float* DPB = (const float*)d_in[6];  // down_proj_bias [8,1024]
  float* out = (float*)d_out;                // [2,512,1024] fp32

  char* ws   = (char*)d_ws;
  int*   cnt  = (int*)ws;                                    // 8 ints
  int*   toks = (int*)(ws + 256);                            // 8*1024 ints
  float* wgts = (float*)(ws + 256 + (size_t)E_EXP * T_TOK * 4);

  moe_zero_kernel<<<(T_TOK * H_DIM + 255) / 256, 256, 0, stream>>>(out, cnt);
  moe_route_kernel<<<(T_TOK + 255) / 256, 256, 0, stream>>>(RI, RW, cnt, toks, wgts);

  (void)hipFuncSetAttribute((const void*)moe_fused_kernel,
                            hipFuncAttributeMaxDynamicSharedMemorySize, SMEM_BYTES);
  dim3 grid(T_TOK / TILE_M, E_EXP);   // worst case: all tokens on one expert
  moe_fused_kernel<<<grid, THREADS, SMEM_BYTES, stream>>>(
      X, GU, GUB, DP, DPB, cnt, toks, wgts, out);
}
